// DLRM_DCN_Net_72121090835005
// MI455X (gfx1250) — compile-verified, matrix operand order checked by reading, not run on hardware
//
#include <hip/hip_runtime.h>

// ---------------------------------------------------------------------------
// DLRM-DCN forward for MI455X (gfx1250, wave32, WMMA f32_16x16x32_f16)
// ---------------------------------------------------------------------------

typedef _Float16 v16h __attribute__((ext_vector_type(16)));
typedef _Float16 v8h  __attribute__((ext_vector_type(8)));
typedef float    v8f  __attribute__((ext_vector_type(8)));

#define B_ROWS   4096
#define T_TAB    26
#define D_EMB    128
#define L_HOT    20
#define ROWS_TAB 100000
#define F_IN     3456   // (T+1)*D

// ---------------------------------------------------------------------------
// fp32 -> fp16 weight conversion (no padding)
// ---------------------------------------------------------------------------
__global__ void cvt_f32_f16(const float* __restrict__ src,
                            _Float16* __restrict__ dst, int n) {
    for (int i = blockIdx.x * blockDim.x + threadIdx.x; i < n;
         i += gridDim.x * blockDim.x)
        dst[i] = (_Float16)src[i];
}

// fp32 [N,K] -> fp16 [N,ldd], zero-padded in K
__global__ void pad_cvt_f16(const float* __restrict__ src,
                            _Float16* __restrict__ dst, int N, int K, int ldd) {
    int total = N * ldd;
    for (int i = blockIdx.x * blockDim.x + threadIdx.x; i < total;
         i += gridDim.x * blockDim.x) {
        int n = i / ldd, k = i - n * ldd;
        dst[i] = (k < K) ? (_Float16)src[(size_t)n * K + k] : (_Float16)0.0f;
    }
}

// fp32 [N,K] -> fp32 [N,ldd], zero-padded in K
__global__ void pad_cvt_f32(const float* __restrict__ src,
                            float* __restrict__ dst, int N, int K, int ldd) {
    int total = N * ldd;
    for (int i = blockIdx.x * blockDim.x + threadIdx.x; i < total;
         i += gridDim.x * blockDim.x) {
        int n = i / ldd, k = i - n * ldd;
        dst[i] = (k < K) ? src[(size_t)n * K + k] : 0.0f;
    }
}

// ---------------------------------------------------------------------------
// WMMA GEMM:  out[M,N] = epilogue( A[M,K] * W[N,K]^T + bias )
//   A: fp32 (lda row stride, K % 32 == 0, lda % 4 == 0), converted to f16
//      while staging to LDS via b128 loads/stores (branch-free)
//   W: fp16 row-major [N,K]
//   epilogue: relu, or fused DCN cross: out = comb*(v+bias) + xlin
// Block: 256 threads (8 waves). Macro tile 128(M) x 128(N), K step 32.
// Wave tile 32(M) x 64(N): 2 x 4 WMMA tiles -> 8 WMMAs per 6 fragment loads.
// Single LDS buffer with register prefetch of the next K-step.
// Requires: M % 128 == 0, N % 128 == 0, K % 32 == 0.
// ---------------------------------------------------------------------------
__global__ __launch_bounds__(256) void gemm_wmma_f16(
    const float* __restrict__ A, const _Float16* __restrict__ W,
    const float* __restrict__ bias, float* __restrict__ out,
    int M, int N, int K, int lda, int ldc, int relu,
    const float* __restrict__ comb, const float* __restrict__ xlin)
{
    __shared__ _Float16 sA[128 * 40];   // 32 cols + 8 pad halves per row
    __shared__ _Float16 sB[128 * 40];

    const int tid  = threadIdx.x;
    const int n0   = blockIdx.x * 128;  // N varies fastest -> A reuse in L2
    const int m0   = blockIdx.y * 128;
    const int wave = tid >> 5, lane = tid & 31;
    const int wm = wave & 3, wn = wave >> 2;   // 4 M-slots x 2 N-slots
    const int lrow = lane & 15, hi = lane >> 4;

    v8f acc[2][4] = {};

    // staging assignments: 128 rows x 2 half-rows of 16 for both A and B
    const int arow = tid >> 1, acb = (tid & 1) * 16;
    const int brow = tid >> 1, bcb = (tid & 1) * 16;

    const float*    gA = A + (size_t)(m0 + arow) * lda + acb;
    const _Float16* gB = W + (size_t)(n0 + brow) * K + bcb;

    // prefetch k0 = 0
    float4 ra0 = ((const float4*)gA)[0];
    float4 ra1 = ((const float4*)gA)[1];
    float4 ra2 = ((const float4*)gA)[2];
    float4 ra3 = ((const float4*)gA)[3];
    v8h    rb0 = ((const v8h*)gB)[0];
    v8h    rb1 = ((const v8h*)gB)[1];

    for (int k0 = 0; k0 < K; k0 += 32) {
        // ---- commit staged registers to LDS ----
        v8h h0, h1;
        h0[0]=(_Float16)ra0.x; h0[1]=(_Float16)ra0.y; h0[2]=(_Float16)ra0.z; h0[3]=(_Float16)ra0.w;
        h0[4]=(_Float16)ra1.x; h0[5]=(_Float16)ra1.y; h0[6]=(_Float16)ra1.z; h0[7]=(_Float16)ra1.w;
        h1[0]=(_Float16)ra2.x; h1[1]=(_Float16)ra2.y; h1[2]=(_Float16)ra2.z; h1[3]=(_Float16)ra2.w;
        h1[4]=(_Float16)ra3.x; h1[5]=(_Float16)ra3.y; h1[6]=(_Float16)ra3.z; h1[7]=(_Float16)ra3.w;
        *(v8h*)&sA[arow * 40 + acb]     = h0;
        *(v8h*)&sA[arow * 40 + acb + 8] = h1;
        *(v8h*)&sB[brow * 40 + bcb]     = rb0;
        *(v8h*)&sB[brow * 40 + bcb + 8] = rb1;
        __syncthreads();

        // ---- prefetch next K-step (uniform branch, EXEC unchanged) ----
        if (k0 + 32 < K) {
            const float4* nA = (const float4*)(gA + k0 + 32);
            ra0 = nA[0]; ra1 = nA[1]; ra2 = nA[2]; ra3 = nA[3];
            const v8h* nB = (const v8h*)(gB + k0 + 32);
            rb0 = nB[0]; rb1 = nB[1];
        }

        // ---- fragments per ISA 16-bit A/B wave32 layout ----
        // lanes 0-15: K = 0..7 and 16..23; lanes 16-31: K = 8..15 and 24..31
        v16h af[2], bf[4];
        #pragma unroll
        for (int t = 0; t < 2; ++t) {
            const _Float16* pa = &sA[(wm * 32 + t * 16 + lrow) * 40 + hi * 8];
            v8h alo = *(const v8h*)pa;
            v8h ahi = *(const v8h*)(pa + 16);
            af[t] = __builtin_shufflevector(alo, ahi, 0, 1, 2, 3, 4, 5, 6, 7,
                                            8, 9, 10, 11, 12, 13, 14, 15);
        }
        #pragma unroll
        for (int t = 0; t < 4; ++t) {
            const _Float16* pb = &sB[(wn * 64 + t * 16 + lrow) * 40 + hi * 8];
            v8h blo = *(const v8h*)pb;
            v8h bhi = *(const v8h*)(pb + 16);
            bf[t] = __builtin_shufflevector(blo, bhi, 0, 1, 2, 3, 4, 5, 6, 7,
                                            8, 9, 10, 11, 12, 13, 14, 15);
        }

        #pragma unroll
        for (int tm = 0; tm < 2; ++tm)
            #pragma unroll
            for (int tn = 0; tn < 4; ++tn)
                acc[tm][tn] = __builtin_amdgcn_wmma_f32_16x16x32_f16(
                    false, af[tm], false, bf[tn], (short)0, acc[tm][tn],
                    false, false);
        __syncthreads();
    }

    // ---- epilogue: C/D layout VGPR r -> M = r + 8*hi, N = lrow ----
    #pragma unroll
    for (int tn = 0; tn < 4; ++tn) {
        const int gn = n0 + wn * 64 + tn * 16 + lrow;
        const float bv = bias ? bias[gn] : 0.0f;
        #pragma unroll
        for (int tm = 0; tm < 2; ++tm) {
            const int mb = m0 + wm * 32 + tm * 16 + hi * 8;
            #pragma unroll
            for (int r = 0; r < 8; ++r) {
                const size_t o = (size_t)(mb + r) * ldc + gn;
                float v = acc[tm][tn][r] + bv;
                if (relu) v = fmaxf(v, 0.0f);
                if (comb) v = comb[o] * v + xlin[o];   // DCN cross layer
                out[o] = v;
            }
        }
    }
}

// ---------------------------------------------------------------------------
// Multi-table embedding-bag sum. One wave per (table, bag).
// seg-sum semantics of the reference: bag b covers
//   [ b==0 ? 0 : off[b],  b==B-1 ? nt : off[b+1] )
// Each lane owns 4 of the 128 features (float4 = global_load_b128).
// ---------------------------------------------------------------------------
__global__ __launch_bounds__(256) void emb_bag_kernel(
    const float* __restrict__ tables, const int* __restrict__ indices,
    const int* __restrict__ offsets, float* __restrict__ combined)
{
    const int wave = (blockIdx.x * blockDim.x + threadIdx.x) >> 5;
    const int lane = threadIdx.x & 31;
    if (wave >= T_TAB * B_ROWS) return;
    const int t = wave / B_ROWS;
    const int b = wave - t * B_ROWS;
    const int nt = B_ROWS * L_HOT;

    const int* offt = offsets + (size_t)t * B_ROWS;
    const int start = (b == 0) ? 0 : offt[b];
    const int end   = (b == B_ROWS - 1) ? nt : offt[b + 1];

    const int*   idxt = indices + (size_t)t * nt;
    const float* tab  = tables + (size_t)t * ROWS_TAB * D_EMB;

    float4 acc = make_float4(0.f, 0.f, 0.f, 0.f);
    for (int i = start; i < end; ++i) {
        const int r = idxt[i];
        const float4 v = ((const float4*)(tab + (size_t)r * D_EMB))[lane];
        acc.x += v.x; acc.y += v.y; acc.z += v.z; acc.w += v.w;
    }
    float4* dst = (float4*)(combined + (size_t)b * F_IN + (1 + t) * D_EMB);
    dst[lane] = acc;
}

// ---------------------------------------------------------------------------
// Final top layer (N=1): wave-per-row dot product + bias, fp32.
// ---------------------------------------------------------------------------
__global__ __launch_bounds__(256) void top_final_kernel(
    const float* __restrict__ z, const float* __restrict__ w3,
    const float* __restrict__ b3, float* __restrict__ out)
{
    const int row  = (blockIdx.x * blockDim.x + threadIdx.x) >> 5;
    const int lane = threadIdx.x & 31;
    if (row >= B_ROWS) return;
    const float* zr = z + (size_t)row * 512;
    float s = 0.f;
    #pragma unroll
    for (int j = 0; j < 16; ++j) s += zr[lane + 32 * j] * w3[lane + 32 * j];
    #pragma unroll
    for (int o = 16; o > 0; o >>= 1) s += __shfl_down(s, o, 32);
    if (lane == 0) out[row] = s + b3[0];
}

// ---------------------------------------------------------------------------
extern "C" void kernel_launch(void* const* d_in, const int* in_sizes, int n_in,
                              void* d_out, int out_size, void* d_ws,
                              size_t ws_size, hipStream_t stream)
{
    (void)in_sizes; (void)n_in; (void)out_size; (void)ws_size;

    const float* dense   = (const float*)d_in[0];
    const int*   indices = (const int*)d_in[1];
    const int*   offsets = (const int*)d_in[2];
    const float* tables  = (const float*)d_in[3];
    const float* botW0 = (const float*)d_in[4];  const float* botb0 = (const float*)d_in[5];
    const float* botW1 = (const float*)d_in[6];  const float* botb1 = (const float*)d_in[7];
    const float* botW2 = (const float*)d_in[8];  const float* botb2 = (const float*)d_in[9];
    const float* dcnW  = (const float*)d_in[10];
    const float* dcnV  = (const float*)d_in[11];
    const float* dcnb  = (const float*)d_in[12];
    const float* topW0 = (const float*)d_in[13]; const float* topb0 = (const float*)d_in[14];
    const float* topW1 = (const float*)d_in[15]; const float* topb1 = (const float*)d_in[16];
    const float* topW2 = (const float*)d_in[17]; const float* topb2 = (const float*)d_in[18];
    const float* topW3 = (const float*)d_in[19]; const float* topb3 = (const float*)d_in[20];
    float* out = (float*)d_out;

    // ---- carve workspace (256B aligned) ----
    char* p = (char*)d_ws;
    auto carve = [&](size_t bytes) -> void* {
        void* r = (void*)p;
        p += (bytes + 255) & ~(size_t)255;
        return r;
    };
    float* combined  = (float*)carve((size_t)B_ROWS * F_IN * 4);
    float* xl        = (float*)carve((size_t)B_ROWS * F_IN * 4);
    float* t1        = (float*)carve((size_t)B_ROWS * 512 * 4);   // also z2
    float* scrA      = (float*)carve((size_t)B_ROWS * 1024 * 4);  // h0 / z0
    float* scrB      = (float*)carve((size_t)B_ROWS * 1024 * 4);  // h1 / z1
    float* densePad  = (float*)carve((size_t)B_ROWS * 32 * 4);    // K 13 -> 32
    _Float16* wBot0 = (_Float16*)carve((size_t)512 * 32 * 2);     // padded
    _Float16* wBot1 = (_Float16*)carve((size_t)256 * 512 * 2);
    _Float16* wBot2 = (_Float16*)carve((size_t)128 * 256 * 2);
    _Float16* wDcnV = (_Float16*)carve((size_t)3 * 512 * F_IN * 2);
    _Float16* wDcnW = (_Float16*)carve((size_t)3 * F_IN * 512 * 2);
    _Float16* wTop0 = (_Float16*)carve((size_t)1024 * F_IN * 2);
    _Float16* wTop1 = (_Float16*)carve((size_t)1024 * 1024 * 2);
    _Float16* wTop2 = (_Float16*)carve((size_t)512 * 1024 * 2);

    auto cvt = [&](const float* s, _Float16* d, int n) {
        int blocks = (n + 255) / 256;
        if (blocks > 4096) blocks = 4096;
        cvt_f32_f16<<<dim3(blocks), dim3(256), 0, stream>>>(s, d, n);
    };
    // padded conversions for the K=13 layer
    pad_cvt_f32<<<dim3(512), dim3(256), 0, stream>>>(dense, densePad,
                                                     B_ROWS, 13, 32);
    pad_cvt_f16<<<dim3(64), dim3(256), 0, stream>>>(botW0, wBot0, 512, 13, 32);
    cvt(botW1, wBot1, 256 * 512);
    cvt(botW2, wBot2, 128 * 256);
    cvt(dcnV,  wDcnV, 3 * 512 * F_IN);
    cvt(dcnW,  wDcnW, 3 * F_IN * 512);
    cvt(topW0, wTop0, 1024 * F_IN);
    cvt(topW1, wTop1, 1024 * 1024);
    cvt(topW2, wTop2, 512 * 1024);

    auto gemm = [&](const float* A, const _Float16* W, const float* bias,
                    float* o, int N, int K, int lda, int ldc, int relu,
                    const float* comb, const float* xlin) {
        dim3 grid(N / 128, B_ROWS / 128);
        gemm_wmma_f16<<<grid, dim3(256), 0, stream>>>(
            A, W, bias, o, B_ROWS, N, K, lda, ldc, relu, comb, xlin);
    };

    // ---- bottom MLP (ReLU each layer); last layer writes combined cols 0..127
    gemm(densePad, wBot0, botb0, scrA,     512, 32,  32,   512,  1, nullptr, nullptr);
    gemm(scrA,     wBot1, botb1, scrB,     256, 512, 512,  256,  1, nullptr, nullptr);
    gemm(scrB,     wBot2, botb2, combined, 128, 256, 256,  F_IN, 1, nullptr, nullptr);

    // ---- embedding bags -> combined cols 128..3455 ----
    {
        const int nwaves = T_TAB * B_ROWS;               // 106496
        const int blocks = (nwaves + 7) / 8;             // 8 waves / block
        emb_bag_kernel<<<dim3(blocks), dim3(256), 0, stream>>>(
            tables, indices, offsets, combined);
    }

    // ---- DCN low-rank cross layers (fused elementwise epilogue) ----
    gemm(combined, wDcnV, nullptr, t1, 512, F_IN, F_IN, 512, 0, nullptr, nullptr);
    gemm(t1, wDcnW, dcnb, xl, F_IN, 512, 512, F_IN, 0, combined, combined);
    for (int l = 1; l < 3; ++l) {
        const _Float16* Vl = wDcnV + (size_t)l * 512 * F_IN;
        const _Float16* Wl = wDcnW + (size_t)l * F_IN * 512;
        const float*    bl = dcnb + (size_t)l * F_IN;
        gemm(xl, Vl, nullptr, t1, 512, F_IN, F_IN, 512, 0, nullptr, nullptr);
        gemm(t1, Wl, bl, xl, F_IN, 512, 512, F_IN, 0, combined, xl);
    }

    // ---- top MLP ----
    gemm(xl,   wTop0, topb0, scrA, 1024, F_IN, F_IN, 1024, 1, nullptr, nullptr);
    gemm(scrA, wTop1, topb1, scrB, 1024, 1024, 1024, 1024, 1, nullptr, nullptr);
    gemm(scrB, wTop2, topb2, t1,   512,  1024, 1024, 512,  1, nullptr, nullptr);

    // ---- final N=1 layer ----
    top_final_kernel<<<dim3(B_ROWS / 8), dim3(256), 0, stream>>>(
        t1, topW3, topb3, out);
}
